// Decoder_9148280340615
// MI455X (gfx1250) — compile-verified, matrix operand order checked by reading, not run on hardware
//
#include <hip/hip_runtime.h>
#include <math.h>

typedef __bf16 bfloat;
typedef __attribute__((ext_vector_type(16))) __bf16 v16bf;
typedef __attribute__((ext_vector_type(8)))  __bf16 v8bf;
typedef __attribute__((ext_vector_type(8)))  float  v8f;

#define BATCH 32
#define TENC 512
#define NSTEP 501
#define TOUT 1002
#define TPAD 1006

__device__ __forceinline__ float sigf(float x) { return 1.f / (1.f + __expf(-x)); }

__device__ __forceinline__ v8f wmma_bf16(v16bf a, v16bf b, v8f c) {
    return __builtin_amdgcn_wmma_f32_16x16x32_bf16(false, a, false, b, (short)0, c, false, false);
}

// ---------------- Weight repack: fp32 [N,K] (strided) -> bf16 WMMA B-blocks ----------
// dst block layout: [nt][ktOff+kt][512], element = lane*16 + j
// lane<16: n=nt*16+lane, j<8 -> k=kt*32+j, j>=8 -> k=kt*32+16+(j-8); lane>=16: +8 on k
__global__ void repack_k(bfloat* __restrict__ dst, const float* __restrict__ src,
                         int rowStride, int elemStride, int Ksrc, int ktOff, int KTtot) {
    int nt = blockIdx.x, kt = blockIdx.y, lane = threadIdx.x;
    int n = nt * 16 + (lane & 15);
    int kbase = kt * 32 + ((lane & 16) ? 8 : 0);
    bfloat* o = dst + ((size_t)nt * KTtot + ktOff + kt) * 512 + lane * 16;
#pragma unroll
    for (int j = 0; j < 16; ++j) {
        int k = kbase + (j < 8 ? j : 8 + j);
        float v = (k < Ksrc) ? src[(size_t)n * rowStride + (size_t)k * elemStride] : 0.f;
        o[j] = (bfloat)v;
    }
}

// ---------------- GEMM, M=32 fixed, NTW n-tiles per wave ----------------------------
// out[32,N] = A[32,K](bf16) @ W^T + bias
template <int NTW>
__global__ void __launch_bounds__(32)
gemm_m32_t(const bfloat* __restrict__ A, int lda,
           const bfloat* __restrict__ Wp, int KT,
           const float* __restrict__ bias,
           float* __restrict__ out, int ldo) {
    int ntBase = blockIdx.x * NTW;
    int lane = threadIdx.x;
    int lr = lane & 15;
    int klo = (lane & 16) ? 8 : 0;
    const bfloat* arow0 = A + (size_t)lr * lda + klo;
    const bfloat* arow1 = A + (size_t)(16 + lr) * lda + klo;
    const bfloat* bblk = Wp + (size_t)ntBase * KT * 512 + lane * 16;
    v8f c0[NTW], c1[NTW];
#pragma unroll
    for (int j = 0; j < NTW; ++j) { c0[j] = (v8f){}; c1[j] = (v8f){}; }
    for (int kt = 0; kt < KT; ++kt) {
        v8bf alo0 = *(const v8bf*)(arow0);
        v8bf ahi0 = *(const v8bf*)(arow0 + 16);
        v8bf alo1 = *(const v8bf*)(arow1);
        v8bf ahi1 = *(const v8bf*)(arow1 + 16);
        arow0 += 32; arow1 += 32;
        v16bf a0, a1;
#pragma unroll
        for (int i = 0; i < 8; ++i) { a0[i] = alo0[i]; a0[i + 8] = ahi0[i];
                                      a1[i] = alo1[i]; a1[i + 8] = ahi1[i]; }
#pragma unroll
        for (int j = 0; j < NTW; ++j) {
            v16bf b = *(const v16bf*)(bblk + ((size_t)j * KT + kt) * 512);
            c0[j] = wmma_bf16(a0, b, c0[j]);
            c1[j] = wmma_bf16(a1, b, c1[j]);
        }
    }
    int r0 = (lane & 16) ? 8 : 0;
#pragma unroll
    for (int j = 0; j < NTW; ++j) {
        int col = (ntBase + j) * 16 + lr;
        float bv = bias[col];
#pragma unroll
        for (int r = 0; r < 8; ++r) {
            out[(size_t)(r0 + r) * ldo + col] = c0[j][r] + bv;
            out[(size_t)(16 + r0 + r) * ldo + col] = c1[j][r] + bv;
        }
    }
}

// ---------------- Postnet conv-as-GEMM with fused BN+tanh epilogue ------------------
// Apad: bf16 (32, 1006, Cin); out (mode 0): bf16 (32, 1006, CoutPad) interior rows
// mode 1 (final layer): dmels[b,t,c] = mask ? 0 : mels_pre + tanh(bn(conv))
template <int NTW>
__global__ void __launch_bounds__(32)
gemm_conv_t(const bfloat* __restrict__ Apad, int Cin,
            const bfloat* __restrict__ Wp, int KT, int NT,
            const float* __restrict__ cb,
            const float* __restrict__ gamma, const float* __restrict__ beta,
            const float* __restrict__ mean, const float* __restrict__ var,
            bfloat* __restrict__ outPad, int CoutPad,
            int mode, const float* __restrict__ mels_pre,
            const unsigned char* __restrict__ mmask,
            float* __restrict__ dmels) {
    int mt = blockIdx.x;
    int ntBase = blockIdx.y * NTW;
    int lane = threadIdx.x;
    int lr = lane & 15;
    int klo = (lane & 16) ? 8 : 0;
    int m = mt * 16 + lr;
    int b = m / TOUT, t = m % TOUT;
    const bfloat* arow_base = Apad + ((size_t)b * TPAD + t) * Cin + klo;
    size_t tapStride = (size_t)NT * KT * 512;
    v8f c[NTW];
#pragma unroll
    for (int j = 0; j < NTW; ++j) c[j] = (v8f){};
    for (int k = 0; k < 5; ++k) {
        const bfloat* ar = arow_base + (size_t)k * Cin;
        const bfloat* bb = Wp + (size_t)k * tapStride + (size_t)ntBase * KT * 512 + lane * 16;
        for (int kt = 0; kt < KT; ++kt) {
            v8bf alo = *(const v8bf*)(ar);
            v8bf ahi = *(const v8bf*)(ar + 16);
            ar += 32;
            v16bf av;
#pragma unroll
            for (int i = 0; i < 8; ++i) { av[i] = alo[i]; av[i + 8] = ahi[i]; }
#pragma unroll
            for (int j = 0; j < NTW; ++j) {
                v16bf bv = *(const v16bf*)(bb + ((size_t)j * KT + kt) * 512);
                c[j] = wmma_bf16(av, bv, c[j]);
            }
        }
    }
    int r0 = (lane & 16) ? 8 : 0;
#pragma unroll
    for (int j = 0; j < NTW; ++j) {
        int col = (ntBase + j) * 16 + lr;
        float bnb = cb[col], g = gamma[col], be = beta[col], mu = mean[col];
        float inv = rsqrtf(var[col] + 1e-5f);
#pragma unroll
        for (int r = 0; r < 8; ++r) {
            int mr = mt * 16 + r0 + r;
            int rb = mr / TOUT, rt = mr % TOUT;
            float v = (c[j][r] + bnb - mu) * inv * g + be;
            v = tanhf(v);
            if (mode == 0) {
                outPad[((size_t)rb * TPAD + rt + 2) * CoutPad + col] = (bfloat)v;
            } else {
                float mp = mels_pre[((size_t)(rt >> 1) * BATCH + rb) * 160 + (rt & 1) * 80 + col];
                dmels[((size_t)rb * TOUT + rt) * 80 + col] = mmask[rb] ? 0.f : (mp + v);
            }
        }
    }
}

// ---------------- Small VALU kernels -----------------------------------------------
__global__ void vecadd_k(float* o, const float* a, const float* b, int n) {
    int i = blockIdx.x * blockDim.x + threadIdx.x;
    if (i < n) o[i] = a[i] + b[i];
}

__global__ void prenet1_k(const float* __restrict__ mels, const float* __restrict__ w,
                          const float* __restrict__ bias, float* __restrict__ out1) {
    int row = blockIdx.x;                 // t*32 + b, t in [0,501)
    int t = row / BATCH, b = row % BATCH;
    int o = threadIdx.x;                  // 256
    float acc = bias[o];
    if (t > 0) {
        const float* x = mels + (size_t)b * 80000 + (size_t)(t - 1) * 160;
        const float* wr = w + (size_t)o * 160;
        for (int k = 0; k < 160; ++k) acc += wr[k] * x[k];
    }
    out1[(size_t)row * 256 + o] = fmaxf(acc, 0.f);
}

__global__ void prenet2_k(const float* __restrict__ out1, const float* __restrict__ w,
                          const float* __restrict__ bias, bfloat* __restrict__ Xpre) {
    int row = blockIdx.x;
    int o = threadIdx.x;
    const float* x = out1 + (size_t)row * 256;
    const float* wr = w + (size_t)o * 256;
    float acc = bias[o];
    for (int k = 0; k < 256; ++k) acc += wr[k] * x[k];
    Xpre[(size_t)row * 256 + o] = (bfloat)fmaxf(acc, 0.f);
}

__global__ void memkey_k(const float* __restrict__ enc, const float* __restrict__ w,
                         const float* __restrict__ bias, float* __restrict__ mk) {
    int row = blockIdx.x;   // b*512 + pos
    int o = threadIdx.x;    // 128
    const float* x = enc + (size_t)row * 512;
    const float* wr = w + (size_t)o * 512;
    float acc = bias[o];
    for (int k = 0; k < 512; ++k) acc += wr[k] * x[k];
    mk[(size_t)row * 128 + o] = acc;
}

__global__ void copyx_k(bfloat* __restrict__ actAtt, const bfloat* __restrict__ Xt) {
    int i = blockIdx.x * blockDim.x + threadIdx.x;   // 32*256
    int b = i >> 8, c = i & 255;
    actAtt[(size_t)b * 1792 + c] = Xt[i];
}

__global__ void lstm_pw(const float* __restrict__ gates, float* __restrict__ cst,
                        float* __restrict__ hst,
                        bfloat* __restrict__ hslot1, int stride1,
                        bfloat* __restrict__ hslot2, int stride2) {
    int i = blockIdx.x * blockDim.x + threadIdx.x;   // 32*1024
    int b = i >> 10, u = i & 1023;
    const float* g = gates + (size_t)b * 4096;
    float ig = g[u], fg = g[1024 + u], gg = g[2048 + u], og = g[3072 + u];
    float c = sigf(fg) * cst[i] + sigf(ig) * tanhf(gg);
    float h = sigf(og) * tanhf(c);
    cst[i] = c; hst[i] = h;
    bfloat hb = (bfloat)h;
    hslot1[(size_t)b * stride1 + u] = hb;
    hslot2[(size_t)b * stride2 + u] = hb;
}

__global__ void att_e_k(const float* __restrict__ aw, const float* __restrict__ awt,
                        const float* __restrict__ query, const float* __restrict__ mk,
                        const float* __restrict__ convw, const float* __restrict__ convb,
                        const float* __restrict__ llw, const float* __restrict__ llb,
                        const float* __restrict__ attw, const float* __restrict__ attb,
                        const unsigned char* __restrict__ tmask, float* __restrict__ e) {
    int blk = blockIdx.x;            // b*512 + pos
    int b = blk >> 9, pos = blk & 511;
    int tid = threadIdx.x;           // 128
    __shared__ float ch[32];
    __shared__ float red[128];
    if (tid < 32) {
        float acc = convb[tid];
        const float* w0 = convw + tid * 62;
        const float* w1 = w0 + 31;
        const float* ar = aw + b * 512;
        const float* atr = awt + b * 512;
        for (int d = 0; d < 31; ++d) {
            int p = pos + d - 15;
            if (p >= 0 && p < 512) acc += w0[d] * ar[p] + w1[d] * atr[p];
        }
        ch[tid] = acc;
    }
    __syncthreads();
    float lf = llb[tid];
    const float* lw = llw + tid * 32;
#pragma unroll
    for (int j = 0; j < 32; ++j) lf += lw[j] * ch[j];
    red[tid] = tanhf(query[b * 128 + tid] + mk[(size_t)blk * 128 + tid] + lf) * attw[tid];
    __syncthreads();
    for (int s = 64; s > 0; s >>= 1) { if (tid < s) red[tid] += red[tid + s]; __syncthreads(); }
    if (tid == 0) {
        float ev = red[0] + attb[0];
        if (tmask[b * 512 + pos]) ev = -INFINITY;
        e[blk] = ev;
    }
}

__global__ void softmax_k(const float* __restrict__ e, float* __restrict__ aw,
                          float* __restrict__ awt, float* __restrict__ dout_att,
                          int t, const unsigned char* __restrict__ mmask) {
    int b = blockIdx.x; int tid = threadIdx.x;  // 512
    __shared__ float red[512];
    float ev = e[b * 512 + tid];
    red[tid] = ev; __syncthreads();
    for (int s = 256; s > 0; s >>= 1) { if (tid < s) red[tid] = fmaxf(red[tid], red[tid + s]); __syncthreads(); }
    float mx = red[0]; __syncthreads();
    float ex = __expf(ev - mx);
    red[tid] = ex; __syncthreads();
    for (int s = 256; s > 0; s >>= 1) { if (tid < s) red[tid] += red[tid + s]; __syncthreads(); }
    float a = ex / red[0];
    aw[b * 512 + tid] = a;
    awt[b * 512 + tid] += a;
    dout_att[((size_t)b * NSTEP + t) * 512 + tid] = mmask[b] ? 0.f : a;
}

__global__ void ctx_k(const float* __restrict__ aw, const float* __restrict__ enc,
                      float* __restrict__ ctx, bfloat* __restrict__ s1,
                      bfloat* __restrict__ s2, bfloat* __restrict__ s3) {
    int b = blockIdx.x, c = threadIdx.x;   // 512
    __shared__ float a[512];
    a[c] = aw[b * 512 + c]; __syncthreads();
    const float* er = enc + (size_t)b * 512 * 512 + c;
    float acc = 0.f;
    for (int tp = 0; tp < 512; ++tp) acc += a[tp] * er[(size_t)tp * 512];
    ctx[b * 512 + c] = acc;
    bfloat cb = (bfloat)acc;
    s1[(size_t)b * 1792 + 256 + c] = cb;   // act_att ctx slot
    s2[(size_t)b * 2560 + c] = cb;         // act_dec ctx slot
    s3[(size_t)b * 1536 + 1024 + c] = cb;  // act_mel ctx slot
}

__global__ void gate_k(const float* __restrict__ dh, const float* __restrict__ ctx,
                       const float* __restrict__ wg, const float* __restrict__ bg,
                       float* __restrict__ dout_gate, int t,
                       const unsigned char* __restrict__ mmask) {
    int b = blockIdx.x; int tid = threadIdx.x;   // 256
    __shared__ float red[256];
    float acc = 0.f;
    for (int k = tid; k < 1024; k += 256) acc += wg[k] * dh[b * 1024 + k];
    for (int k = tid; k < 512; k += 256) acc += wg[1024 + k] * ctx[b * 512 + k];
    red[tid] = acc; __syncthreads();
    for (int s = 128; s > 0; s >>= 1) { if (tid < s) red[tid] += red[tid + s]; __syncthreads(); }
    if (tid == 0) dout_gate[b * NSTEP + t] = mmask[b] ? 1000.f : (red[0] + bg[0]);
}

__global__ void buildpad_k(const float* __restrict__ mels_pre, bfloat* __restrict__ Apad) {
    int i = blockIdx.x * blockDim.x + threadIdx.x;   // 32*1002*96
    if (i >= BATCH * TOUT * 96) return;
    int c = i % 96; int rest = i / 96; int t = rest % TOUT; int b = rest / TOUT;
    float v = 0.f;
    if (c < 80) v = mels_pre[((size_t)(t >> 1) * BATCH + b) * 160 + (t & 1) * 80 + c];
    Apad[((size_t)b * TPAD + t + 2) * 96 + c] = (bfloat)v;
}

// ---------------- Host side ---------------------------------------------------------
extern "C" void kernel_launch(void* const* d_in, const int* in_sizes, int n_in,
                              void* d_out, int out_size, void* d_ws, size_t ws_size,
                              hipStream_t stream) {
    const float* enc  = (const float*)d_in[0];
    const float* mels = (const float*)d_in[1];
    const unsigned char* tmask = (const unsigned char*)d_in[2];
    const unsigned char* mmask = (const unsigned char*)d_in[3];
    const float* pn1w = (const float*)d_in[4];  const float* pn1b = (const float*)d_in[5];
    const float* pn2w = (const float*)d_in[6];  const float* pn2b = (const float*)d_in[7];
    const float* awih = (const float*)d_in[8];  const float* awhh = (const float*)d_in[9];
    const float* abih = (const float*)d_in[10]; const float* abhh = (const float*)d_in[11];
    const float* qw = (const float*)d_in[12];   const float* qb = (const float*)d_in[13];
    const float* mkw = (const float*)d_in[14];  const float* mkb = (const float*)d_in[15];
    const float* lcw = (const float*)d_in[16];  const float* lcb = (const float*)d_in[17];
    const float* llw = (const float*)d_in[18];  const float* llb = (const float*)d_in[19];
    const float* attw = (const float*)d_in[20]; const float* attb = (const float*)d_in[21];
    const float* dwih = (const float*)d_in[22]; const float* dwhh = (const float*)d_in[23];
    const float* dbih = (const float*)d_in[24]; const float* dbhh = (const float*)d_in[25];
    const float* melw = (const float*)d_in[26]; const float* melb = (const float*)d_in[27];
    const float* gw = (const float*)d_in[28];   const float* gb = (const float*)d_in[29];

    char* W = (char*)d_ws;
    size_t cur = 0;
    auto alloc = [&](size_t bytes) -> char* {
        char* p = W + cur; cur += (bytes + 255) & ~(size_t)255; return p;
    };
    // bf16 repacked weights
    bfloat* WpAtt = (bfloat*)alloc((size_t)256 * 56 * 512 * 2);   // 4096 x 1792
    bfloat* WpDec = (bfloat*)alloc((size_t)256 * 80 * 512 * 2);   // 4096 x 2560
    bfloat* WpQ   = (bfloat*)alloc((size_t)8 * 32 * 512 * 2);     // 128 x 1024
    bfloat* WpMel = (bfloat*)alloc((size_t)10 * 48 * 512 * 2);    // 160 x 1536
    bfloat* WpPost[5];
    int postNT[5] = {32, 32, 32, 32, 5};
    int postKT[5] = {3, 16, 16, 16, 16};
    int postCin[5] = {80, 512, 512, 512, 512};
    for (int l = 0; l < 5; ++l)
        WpPost[l] = (bfloat*)alloc((size_t)5 * postNT[l] * postKT[l] * 512 * 2);
    float* biasAtt = (float*)alloc(4096 * 4);
    float* biasDec = (float*)alloc(4096 * 4);
    float* out1    = (float*)alloc((size_t)NSTEP * BATCH * 256 * 4);
    bfloat* Xpre   = (bfloat*)alloc((size_t)NSTEP * BATCH * 256 * 2);
    float* memk    = (float*)alloc((size_t)BATCH * 512 * 128 * 4);
    bfloat* actAtt = (bfloat*)alloc((size_t)BATCH * 1792 * 2);
    bfloat* actDec = (bfloat*)alloc((size_t)BATCH * 2560 * 2);
    bfloat* actMel = (bfloat*)alloc((size_t)BATCH * 1536 * 2);
    float* gates   = (float*)alloc((size_t)BATCH * 4096 * 4);
    float* query   = (float*)alloc((size_t)BATCH * 128 * 4);
    float* ebuf    = (float*)alloc((size_t)BATCH * 512 * 4);
    float* awbuf   = (float*)alloc((size_t)BATCH * 512 * 4);
    float* awtbuf  = (float*)alloc((size_t)BATCH * 512 * 4);
    float* ctxbuf  = (float*)alloc((size_t)BATCH * 512 * 4);
    float* ah = (float*)alloc((size_t)BATCH * 1024 * 4);
    float* ac = (float*)alloc((size_t)BATCH * 1024 * 4);
    float* dh = (float*)alloc((size_t)BATCH * 1024 * 4);
    float* dc = (float*)alloc((size_t)BATCH * 1024 * 4);
    float* melsPre = (float*)alloc((size_t)NSTEP * BATCH * 160 * 4);
    size_t padBytes = (size_t)BATCH * TPAD * 512 * 2;
    bfloat* bufA = (bfloat*)alloc(padBytes);
    bfloat* bufB = (bfloat*)alloc(padBytes);

    float* dout_mels = (float*)d_out;
    float* dout_gate = dout_mels + (size_t)BATCH * TOUT * 80;
    float* dout_att  = dout_gate + (size_t)BATCH * NSTEP;

    // ---- zero state & activation buffers (ws is poisoned; must re-init each call) --
    hipMemsetAsync(actAtt, 0, (size_t)BATCH * 1792 * 2, stream);
    hipMemsetAsync(actDec, 0, (size_t)BATCH * 2560 * 2, stream);
    hipMemsetAsync(actMel, 0, (size_t)BATCH * 1536 * 2, stream);
    hipMemsetAsync(ah, 0, (size_t)BATCH * 1024 * 4, stream);
    hipMemsetAsync(ac, 0, (size_t)BATCH * 1024 * 4, stream);
    hipMemsetAsync(dh, 0, (size_t)BATCH * 1024 * 4, stream);
    hipMemsetAsync(dc, 0, (size_t)BATCH * 1024 * 4, stream);
    hipMemsetAsync(awbuf, 0, (size_t)BATCH * 512 * 4, stream);
    hipMemsetAsync(awtbuf, 0, (size_t)BATCH * 512 * 4, stream);

    // ---- one-time repacks / precomputes -------------------------------------------
    repack_k<<<dim3(256, 24), 32, 0, stream>>>(WpAtt, awih, 768, 1, 768, 0, 56);
    repack_k<<<dim3(256, 32), 32, 0, stream>>>(WpAtt, awhh, 1024, 1, 1024, 24, 56);
    repack_k<<<dim3(256, 48), 32, 0, stream>>>(WpDec, dwih, 1536, 1, 1536, 0, 80);
    repack_k<<<dim3(256, 32), 32, 0, stream>>>(WpDec, dwhh, 1024, 1, 1024, 48, 80);
    repack_k<<<dim3(8, 32), 32, 0, stream>>>(WpQ, qw, 1024, 1, 1024, 0, 32);
    repack_k<<<dim3(10, 48), 32, 0, stream>>>(WpMel, melw, 1536, 1, 1536, 0, 48);
    for (int l = 0; l < 5; ++l) {
        const float* pw = (const float*)d_in[30 + l * 6];
        for (int k = 0; k < 5; ++k) {
            bfloat* dst = WpPost[l] + (size_t)k * postNT[l] * postKT[l] * 512;
            repack_k<<<dim3(postNT[l], postKT[l]), 32, 0, stream>>>(
                dst, pw + k, postCin[l] * 5, 5, postCin[l], 0, postKT[l]);
        }
    }
    vecadd_k<<<16, 256, 0, stream>>>(biasAtt, abih, abhh, 4096);
    vecadd_k<<<16, 256, 0, stream>>>(biasDec, dbih, dbhh, 4096);

    prenet1_k<<<NSTEP * BATCH, 256, 0, stream>>>(mels, pn1w, pn1b, out1);
    prenet2_k<<<NSTEP * BATCH, 256, 0, stream>>>(out1, pn2w, pn2b, Xpre);
    memkey_k<<<BATCH * 512, 128, 0, stream>>>(enc, mkw, mkb, memk);

    // ---- sequential decoder scan ---------------------------------------------------
    for (int t = 0; t < NSTEP; ++t) {
        copyx_k<<<32, 256, 0, stream>>>(actAtt, Xpre + (size_t)t * BATCH * 256);
        gemm_m32_t<4><<<64, 32, 0, stream>>>(actAtt, 1792, WpAtt, 56, biasAtt, gates, 4096);
        lstm_pw<<<128, 256, 0, stream>>>(gates, ac, ah, actAtt + 768, 1792, actDec + 512, 2560);
        gemm_m32_t<4><<<2, 32, 0, stream>>>(actAtt + 768, 1792, WpQ, 32, qb, query, 128);
        att_e_k<<<BATCH * 512, 128, 0, stream>>>(awbuf, awtbuf, query, memk,
                                                 lcw, lcb, llw, llb, attw, attb, tmask, ebuf);
        softmax_k<<<BATCH, 512, 0, stream>>>(ebuf, awbuf, awtbuf, dout_att, t, mmask);
        ctx_k<<<BATCH, 512, 0, stream>>>(awbuf, enc, ctxbuf, actAtt, actDec, actMel);
        gemm_m32_t<4><<<64, 32, 0, stream>>>(actDec, 2560, WpDec, 80, biasDec, gates, 4096);
        lstm_pw<<<128, 256, 0, stream>>>(gates, dc, dh, actDec + 1536, 2560, actMel, 1536);
        gemm_m32_t<2><<<5, 32, 0, stream>>>(actMel, 1536, WpMel, 48, melb,
                                            melsPre + (size_t)t * BATCH * 160, 160);
        gate_k<<<BATCH, 256, 0, stream>>>(dh, ctxbuf, gw, gb, dout_gate, t, mmask);
    }

    // ---- postnet -------------------------------------------------------------------
    hipMemsetAsync(bufA, 0, padBytes, stream);
    hipMemsetAsync(bufB, 0, padBytes, stream);
    buildpad_k<<<(BATCH * TOUT * 96 + 255) / 256, 256, 0, stream>>>(melsPre, bufA);
    const int MT = (BATCH * TOUT) / 16;   // 2004
    // L0: bufA(96) -> bufB(512)
    gemm_conv_t<4><<<dim3(MT, 8), 32, 0, stream>>>(bufA, 96, WpPost[0], 3, 32,
        (const float*)d_in[31], (const float*)d_in[32], (const float*)d_in[33],
        (const float*)d_in[34], (const float*)d_in[35],
        bufB, 512, 0, nullptr, mmask, nullptr);
    hipMemsetAsync(bufA, 0, padBytes, stream);   // clear stale L0-input bytes in pads
    // L1: bufB -> bufA
    gemm_conv_t<4><<<dim3(MT, 8), 32, 0, stream>>>(bufB, 512, WpPost[1], 16, 32,
        (const float*)d_in[37], (const float*)d_in[38], (const float*)d_in[39],
        (const float*)d_in[40], (const float*)d_in[41],
        bufA, 512, 0, nullptr, mmask, nullptr);
    // L2: bufA -> bufB
    gemm_conv_t<4><<<dim3(MT, 8), 32, 0, stream>>>(bufA, 512, WpPost[2], 16, 32,
        (const float*)d_in[43], (const float*)d_in[44], (const float*)d_in[45],
        (const float*)d_in[46], (const float*)d_in[47],
        bufB, 512, 0, nullptr, mmask, nullptr);
    // L3: bufB -> bufA
    gemm_conv_t<4><<<dim3(MT, 8), 32, 0, stream>>>(bufB, 512, WpPost[3], 16, 32,
        (const float*)d_in[49], (const float*)d_in[50], (const float*)d_in[51],
        (const float*)d_in[52], (const float*)d_in[53],
        bufA, 512, 0, nullptr, mmask, nullptr);
    // L4 (final, fused residual + mask -> d_out mels)
    gemm_conv_t<1><<<dim3(MT, 5), 32, 0, stream>>>(bufA, 512, WpPost[4], 16, 5,
        (const float*)d_in[55], (const float*)d_in[56], (const float*)d_in[57],
        (const float*)d_in[58], (const float*)d_in[59],
        nullptr, 80, 1, melsPre, mmask, dout_mels);
}
